// DeformableKPConvX_5317169512772
// MI455X (gfx1250) — matrix operand order repaired and checked by reading, not compiled, not run on hardware
//
#include <hip/hip_runtime.h>
#include <hip/hip_bf16.h>

typedef _Float16 v16h __attribute__((ext_vector_type(16)));
typedef float    v8f  __attribute__((ext_vector_type(8)));

constexpr int B_    = 2;
constexpr int NP    = 8192;
constexpr int CIN   = 64;    // C
constexpr int DOUT_ = 64;
constexpr int KP_   = 15;    // kernel points
constexpr int H_    = 32;    // neighbors
constexpr int G_    = 3;
constexpr int ICG_  = 32;
constexpr int OCG_  = 32;
constexpr int OFF_  = 16;    // channel shift
constexpr int MID_  = 96;    // G*OCG
constexpr int D3K   = 45;    // 3*K
constexpr int NT    = NP / 16;   // 16-point tiles per batch
constexpr float CONVR  = 0.1f;
constexpr float RINV   = 1.0f / 0.04f;   // 1/RECEPT_R

__device__ __forceinline__ float leakyf(float x) { return x >= 0.f ? x : 0.1f * x; }

__device__ __forceinline__ v8f wmma_f16(v16h a, v16h b, v8f c) {
  // v_wmma_f32_16x16x32_f16: D = A(16x32,f16) * B(32x16,f16) + C(16x16,f32)
  return __builtin_amdgcn_wmma_f32_16x16x32_f16(false, a, false, b, (short)0, c, false, false);
}

// A-matrix (16x32 f16) K index for element i of the per-lane v16h
__device__ __forceinline__ int a_k(int lane, int i) {
  return i + ((i >= 8) ? 8 : 0) + ((lane >= 16) ? 8 : 0);
}
// B-matrix (32x16 f16) K index for element i
__device__ __forceinline__ int b_k(int lane, int i) {
  return i + ((i >= 8) ? 8 : 0) + ((lane >= 16) ? 8 : 0);
}

// ---------------------------------------------------------------------------
// Ball query: one wave32 per query point. Scan all N points in ascending
// index order; ballot+prefix-popcount compacts the first H in-radius indices
// (exactly the reference's sort(where(d2>r2, N, j))[:H] semantics). Pad = N.
// ---------------------------------------------------------------------------
__global__ void ball_query_kernel(const float* __restrict__ SP, int* __restrict__ IDX) {
  int lane = threadIdx.x & 31;
  int wid  = threadIdx.x >> 5;
  int pid  = blockIdx.x * 4 + wid;
  int b = pid / NP, n = pid % NP;
  const float* spb = SP + (size_t)b * 3 * NP;
  float px = spb[n], py = spb[NP + n], pz = spb[2 * NP + n];
  int* out = IDX + ((size_t)b * NP + n) * H_;
  int count = 0;
  const float r2 = CONVR * CONVR;
  for (int m0 = 0; m0 < NP; m0 += 32) {
    int m = m0 + lane;
    float dx = spb[m] - px, dy = spb[NP + m] - py, dz = spb[2 * NP + m] - pz;
    bool pred = (dx * dx + dy * dy + dz * dz) <= r2;
    unsigned mask = (unsigned)__ballot(pred);
    int prefix = __popc(mask & ((1u << lane) - 1u));
    int pos = count + prefix;
    if (pred && pos < H_) out[pos] = m;
    count += __popc(mask);
    if (count >= H_) break;
  }
  if (lane >= count) out[lane] = NP;   // pad
}

// ---------------------------------------------------------------------------
// 1x1 conv GEMM: Y[b,d,n] = act( W[d,:] . actin(X[b,:,n]) + bias[d] )
// One wave computes a 16(row) x 16(col) D tile with K-steps of 32 via WMMA.
// PRE: leaky on input. POST: 0 none, 1 leaky, 2 tanh.
// LAYOUT: 0 -> Y[b][Cout][N] (channel-major), 1 -> Y[b][N][Cout] (point-major)
// ---------------------------------------------------------------------------
template <int PRE, int POST, int LAYOUT, typename OutT>
__global__ void featconv_kernel(const float* __restrict__ X, const float* __restrict__ W,
                                const float* __restrict__ bias, OutT* __restrict__ Y,
                                int Cin, int Cout) {
  int lane = threadIdx.x & 31;
  int t = blockIdx.x;
  int b = t / NT;
  int n0 = (t % NT) * 16;
  int mb = blockIdx.y;
  int mrow = mb * 16 + (lane & 15);
  v8f acc = {};
  for (int k0 = 0; k0 < Cin; k0 += 32) {
    v16h a, bm;
#pragma unroll
    for (int i = 0; i < 16; ++i) {
      int kk = a_k(lane, i);
      a[i] = (mrow < Cout) ? (_Float16)W[(size_t)mrow * Cin + k0 + kk] : (_Float16)0.f;
    }
#pragma unroll
    for (int i = 0; i < 16; ++i) {
      int kk = b_k(lane, i);
      float xv = X[((size_t)b * Cin + (k0 + kk)) * NP + n0 + (lane & 15)];
      if constexpr (PRE) xv = leakyf(xv);
      bm[i] = (_Float16)xv;
    }
    acc = wmma_f16(a, bm, acc);
  }
  int col = n0 + (lane & 15);
#pragma unroll
  for (int r = 0; r < 8; ++r) {
    int rr = mb * 16 + r + ((lane >= 16) ? 8 : 0);
    if (rr < Cout) {
      float v = acc[r] + bias[rr];
      if constexpr (POST == 1) v = leakyf(v);
      else if constexpr (POST == 2) v = tanhf(v);
      if constexpr (LAYOUT == 0) Y[((size_t)b * Cout + rr) * NP + col] = (OutT)v;
      else                       Y[((size_t)b * NP + col) * Cout + rr] = (OutT)v;
    }
  }
}

// ---------------------------------------------------------------------------
// wf[b,n,k,c] = sum_h infl[b,n,h,k] * nb_f[b,n,h,c]
// One wave per point n. Lane = neighbor h. Influence + gathered features are
// staged in LDS, then 4 WMMAs (16k x 32h x 64c) produce the tile (k=15 row
// zero-padded). DEFORM=1 uses per-point deformed kernel points.
// ---------------------------------------------------------------------------
template <int DEFORM>
__global__ void wf_kernel(const float* __restrict__ SP, const int* __restrict__ IDX,
                          const float* __restrict__ KPTS, const float* __restrict__ OFFRAW,
                          const _Float16* __restrict__ F, _Float16* __restrict__ WF) {
  __shared__ float    s_infl[H_ * 16];
  __shared__ _Float16 s_nbf[H_ * DOUT_];
  int lane = threadIdx.x & 31;            // neighbor h
  int pid = blockIdx.x;
  int b = pid / NP, n = pid % NP;
  const float* spb = SP + (size_t)b * 3 * NP;
  float px = spb[n], py = spb[NP + n], pz = spb[2 * NP + n];
  int j = IDX[((size_t)b * NP + n) * H_ + lane];
  bool valid = j < NP;
  int jc = valid ? j : 0;
  float rx = spb[jc] - px, ry = spb[NP + jc] - py, rz = spb[2 * NP + jc] - pz;
#pragma unroll
  for (int k = 0; k < KP_; ++k) {
    float kx = KPTS[k * 3 + 0], ky = KPTS[k * 3 + 1], kz = KPTS[k * 3 + 2];
    if constexpr (DEFORM) {
      kx += CONVR * OFFRAW[((size_t)b * D3K + 0 * KP_ + k) * NP + n];
      ky += CONVR * OFFRAW[((size_t)b * D3K + 1 * KP_ + k) * NP + n];
      kz += CONVR * OFFRAW[((size_t)b * D3K + 2 * KP_ + k) * NP + n];
    }
    float dx = rx - kx, dy = ry - ky, dz = rz - kz;
    float sq = dx * dx + dy * dy + dz * dz;
    float f = fmaxf(1.f - sqrtf(fmaxf(sq, 1e-12f)) * RINV, 0.f);
    s_infl[lane * 16 + k] = valid ? f : 0.f;
  }
  s_infl[lane * 16 + 15] = 0.f;           // k pad row
  const _Float16* fc = F + ((size_t)b * NP + jc) * DOUT_;
#pragma unroll 8
  for (int c = 0; c < DOUT_; ++c)
    s_nbf[lane * DOUT_ + c] = valid ? fc[c] : (_Float16)0.f;
  __syncthreads();
  v16h a;   // A[m=k][K=h] = infl[h,k]
#pragma unroll
  for (int i = 0; i < 16; ++i) {
    int h = a_k(lane, i);
    a[i] = (_Float16)s_infl[h * 16 + (lane & 15)];
  }
  _Float16* wout = WF + ((size_t)b * NP + n) * 16 * DOUT_;
#pragma unroll
  for (int cb = 0; cb < 4; ++cb) {
    v16h bm;  // B[K=h][n=c]
#pragma unroll
    for (int i = 0; i < 16; ++i) {
      int h = b_k(lane, i);
      bm[i] = s_nbf[h * DOUT_ + cb * 16 + (lane & 15)];
    }
    v8f acc = {};
    acc = wmma_f16(a, bm, acc);
#pragma unroll
    for (int r = 0; r < 8; ++r) {
      int krow = r + ((lane >= 16) ? 8 : 0);
      wout[krow * DOUT_ + cb * 16 + (lane & 15)] = (_Float16)acc[r];
    }
  }
}

// ---------------------------------------------------------------------------
// out[b, g*32+d, n] = sum_{k,c} wf[b,n,k, g*16+c] * W[k,g,c,d]
// One wave per 16-point tile: per group, 15 K-steps (ICG=32) x 2 N-blocks.
// ---------------------------------------------------------------------------
__global__ void group_contract_kernel(const _Float16* __restrict__ WF,
                                      const float* __restrict__ DW,
                                      float* __restrict__ OUT) {
  int lane = threadIdx.x & 31;
  int t = blockIdx.x;
  int b = t / NT;
  int n0 = (t % NT) * 16;
  int pA = n0 + (lane & 15);              // A row = point
  const _Float16* wfb = WF + ((size_t)b * NP + pA) * 16 * DOUT_;
  for (int g = 0; g < G_; ++g) {
    v8f acc0 = {}, acc1 = {};
    for (int ks = 0; ks < KP_; ++ks) {
      v16h a, b0, b1;
#pragma unroll
      for (int i = 0; i < 16; ++i) {
        int c = a_k(lane, i);             // channel within group window
        a[i] = wfb[ks * DOUT_ + g * OFF_ + c];
      }
#pragma unroll
      for (int i = 0; i < 16; ++i) {
        int c = b_k(lane, i);
        const float* base = DW + ((size_t)(ks * G_ + g) * ICG_ + c) * OCG_;
        b0[i] = (_Float16)base[lane & 15];
        b1[i] = (_Float16)base[16 + (lane & 15)];
      }
      acc0 = wmma_f16(a, b0, acc0);
      acc1 = wmma_f16(a, b1, acc1);
    }
#pragma unroll
    for (int r = 0; r < 8; ++r) {
      int p = n0 + r + ((lane >= 16) ? 8 : 0);
      OUT[((size_t)b * MID_ + g * OCG_ + (lane & 15)) * NP + p]      = acc0[r];
      OUT[((size_t)b * MID_ + g * OCG_ + 16 + (lane & 15)) * NP + p] = acc1[r];
    }
  }
}

extern "C" void kernel_launch(void* const* d_in, const int* in_sizes, int n_in,
                              void* d_out, int out_size, void* d_ws, size_t ws_size,
                              hipStream_t stream) {
  (void)in_sizes; (void)n_in; (void)out_size; (void)ws_size;
  const float* sp    = (const float*)d_in[0];
  const float* sf    = (const float*)d_in[1];
  const float* kp    = (const float*)d_in[2];
  const float* wts   = (const float*)d_in[3];
  const float* dwts  = (const float*)d_in[4];
  const float* dbc_w = (const float*)d_in[5];
  const float* dbc_b = (const float*)d_in[6];
  const float* dec_w = (const float*)d_in[7];
  const float* dec_b = (const float*)d_in[8];
  const float* dcv_w = (const float*)d_in[9];
  const float* dcv_b = (const float*)d_in[10];
  const float* bc_w  = (const float*)d_in[11];
  const float* bc_b  = (const float*)d_in[12];
  const float* ec_w  = (const float*)d_in[13];
  const float* ec_b  = (const float*)d_in[14];
  float* out = (float*)d_out;

  char* ws = (char*)d_ws;
  size_t off = 0;
  auto alloc = [&](size_t bytes) -> char* {
    char* p = ws + off;
    off = (off + bytes + 255) & ~(size_t)255;
    return p;
  };
  int*      idx   = (int*)     alloc((size_t)B_ * NP * H_ * 4);
  _Float16* F1h   = (_Float16*)alloc((size_t)B_ * NP * DOUT_ * 2);   // dbc(sf), point-major
  _Float16* F2h   = (_Float16*)alloc((size_t)B_ * NP * DOUT_ * 2);   // bc(sf), point-major
  _Float16* wfbuf = (_Float16*)alloc((size_t)B_ * NP * 16 * DOUT_ * 2); // reused both branches
  float*    mid   = (float*)   alloc((size_t)B_ * MID_ * NP * 4);    // reused both branches
  float*    offf  = (float*)   alloc((size_t)B_ * CIN * NP * 4);
  float*    offr  = (float*)   alloc((size_t)B_ * D3K * NP * 4);

  dim3 t32(32), t128(128);
  // 1) neighbor indices
  ball_query_kernel<<<dim3(B_ * NP / 4), t128, 0, stream>>>(sp, idx);
  // 2) pointwise feature convs for both branches (f16, point-major for gathers)
  featconv_kernel<0, 0, 1, _Float16><<<dim3(B_ * NT, 4), t32, 0, stream>>>(sf, dbc_w, dbc_b, F1h, CIN, DOUT_);
  featconv_kernel<0, 0, 1, _Float16><<<dim3(B_ * NT, 4), t32, 0, stream>>>(sf, bc_w, bc_b, F2h, CIN, DOUT_);
  // 3) deformation branch
  wf_kernel<0><<<dim3(B_ * NP), t32, 0, stream>>>(sp, idx, kp, nullptr, F1h, wfbuf);
  group_contract_kernel<<<dim3(B_ * NT), t32, 0, stream>>>(wfbuf, dwts, mid);
  featconv_kernel<1, 1, 0, float><<<dim3(B_ * NT, 4), t32, 0, stream>>>(mid, dec_w, dec_b, offf, MID_, CIN);
  featconv_kernel<0, 2, 0, float><<<dim3(B_ * NT, 3), t32, 0, stream>>>(offf, dcv_w, dcv_b, offr, CIN, D3K);
  // 4) main KPConv with deformed kernel points
  wf_kernel<1><<<dim3(B_ * NP), t32, 0, stream>>>(sp, idx, kp, offr, F2h, wfbuf);
  group_contract_kernel<<<dim3(B_ * NT), t32, 0, stream>>>(wfbuf, wts, mid);
  featconv_kernel<1, 1, 0, float><<<dim3(B_ * NT, 4), t32, 0, stream>>>(mid, ec_w, ec_b, out, MID_, DOUT_);
}
